// WindowAttention3D_42880953483909
// MI455X (gfx1250) — compile-verified, matrix operand order checked by reading, not run on hardware
//
#include <hip/hip_runtime.h>
#include <math.h>

typedef __bf16 bf16;
typedef bf16  v8bf  __attribute__((ext_vector_type(8)));
typedef bf16  v16bf __attribute__((ext_vector_type(16)));
typedef float v8f   __attribute__((ext_vector_type(8)));
typedef float v4f   __attribute__((ext_vector_type(4)));

#define LDK 56    // padded stride (elems) for 32-wide bf16 LDS tiles: 112B/row, conflict-free
#define LDW 520   // padded stride (elems) for 512-wide bf16 LDS rows: 1040B/row, conflict-free

// ---- WMMA helpers ------------------------------------------------------
// A/B 16x32 bf16 tile loader matching the CDNA5 16-bit A layout:
// lanes 0-15 hold rows 0-15, K = {0..7, 16..23}; lanes 16-31 same rows, K = {8..15, 24..31}.
static __device__ __forceinline__ v16bf load_tile(const bf16* t, int ld) {
  const int lane = threadIdx.x & 31;
  const bf16* p = t + (lane & 15) * ld + ((lane >> 4) << 3);
  v8bf lo = *(const v8bf*)(p);
  v8bf hi = *(const v8bf*)(p + 16);
  return __builtin_shufflevector(lo, hi, 0,1,2,3,4,5,6,7,8,9,10,11,12,13,14,15);
}

static __device__ __forceinline__ v8f wmma_bf16(v16bf a, v16bf b, v8f c) {
  return __builtin_amdgcn_wmma_f32_16x16x32_bf16(false, a, false, b, (short)0, c, false, false);
}

// ---- small elementwise kernels ----------------------------------------
__global__ void cvt_bf16_kernel(const float* __restrict__ s, bf16* __restrict__ d, int n) {
  int i = blockIdx.x * blockDim.x + threadIdx.x;
  int stride = gridDim.x * blockDim.x;
  for (; i < n; i += stride) d[i] = (bf16)s[i];
}

// CPB MLP: sig16[t][h] = 16*sigmoid( relu(table[t]@w1^T + b1) @ w2^T )[h]
__global__ void cpb_mlp_kernel(const float* __restrict__ table,
                               const float* __restrict__ w1,
                               const float* __restrict__ b1,
                               const float* __restrict__ w2,
                               float* __restrict__ sig16, int T) {
  int t = blockIdx.x * blockDim.x + threadIdx.x;
  if (t >= T) return;
  float x0 = table[t*3+0], x1 = table[t*3+1], x2 = table[t*3+2];
  float acc[8] = {0.f,0.f,0.f,0.f,0.f,0.f,0.f,0.f};
  for (int j = 0; j < 512; ++j) {
    float hv = fmaxf(w1[j*3+0]*x0 + w1[j*3+1]*x1 + w1[j*3+2]*x2 + b1[j], 0.f);
    #pragma unroll
    for (int h = 0; h < 8; ++h) acc[h] += hv * w2[h*512 + j];
  }
  #pragma unroll
  for (int h = 0; h < 8; ++h) sig16[t*8 + h] = 16.f / (1.f + __expf(-acc[h]));
}

// Materialize rpb[h][n][m] = sig16[idx[n][m]][h]  (8 MB, L2 resident)
__global__ void rpb_mat_kernel(const int* __restrict__ idx,
                               const float* __restrict__ sig16,
                               float* __restrict__ rpb) {
  int nm = blockIdx.x * 256 + threadIdx.x;   // 512*512 = 262144
  int id = idx[nm];
  #pragma unroll
  for (int h = 0; h < 8; ++h) rpb[(size_t)h*262144 + nm] = sig16[id*8 + h];
}

// ---- K=256 WMMA GEMM: out = A(Mx256) @ W(Ox256)^T ----------------------
// mode 0: O=768, epilogue adds (q_bias,0,v_bias) and scatters bf16 into
//         q/k/v in (b, h, n, d) layout for the attention kernel.
// mode 1: O=256, epilogue adds proj_b and writes fp32 (token, o) to d_out.
__global__ __launch_bounds__(256, 1)
void gemm_k256_kernel(const bf16* __restrict__ A, const bf16* __restrict__ W,
                      const float* __restrict__ q_bias, const float* __restrict__ v_bias,
                      bf16* __restrict__ qb, bf16* __restrict__ kb, bf16* __restrict__ vb,
                      const float* __restrict__ proj_b, float* __restrict__ out, int mode) {
  __shared__ bf16 Asm[128 * LDK];
  __shared__ bf16 Bsm[128 * LDK];
  const int tid = threadIdx.x;
  const int lane = tid & 31;
  const int wid = tid >> 5;
  const int wm = wid & 1;        // M half (64 rows)
  const int wo = wid >> 1;       // O quarter (32 cols)
  const int tokbase = blockIdx.x * 128;
  const int obase   = blockIdx.y * 128;

  v8f acc[4][2];
  #pragma unroll
  for (int i = 0; i < 4; ++i)
    #pragma unroll
    for (int j = 0; j < 2; ++j) acc[i][j] = (v8f){0.f,0.f,0.f,0.f,0.f,0.f,0.f,0.f};

  const int rstage = tid >> 1;
  const int kstage = (tid & 1) * 16;
  const bf16* ga = A + (size_t)(tokbase + rstage) * 256 + kstage;
  const bf16* gw = W + (size_t)(obase  + rstage) * 256 + kstage;

  for (int kk = 0; kk < 256; kk += 32) {
    __syncthreads();
    {
      v8bf a0 = *(const v8bf*)(ga + kk);
      v8bf a1 = *(const v8bf*)(ga + kk + 8);
      v8bf w0 = *(const v8bf*)(gw + kk);
      v8bf w1 = *(const v8bf*)(gw + kk + 8);
      *(v8bf*)(Asm + rstage*LDK + kstage)     = a0;
      *(v8bf*)(Asm + rstage*LDK + kstage + 8) = a1;
      *(v8bf*)(Bsm + rstage*LDK + kstage)     = w0;
      *(v8bf*)(Bsm + rstage*LDK + kstage + 8) = w1;
      if (kk + 32 < 256) {                    // global_prefetch_b8 next K-chunk
        __builtin_prefetch(ga + kk + 32, 0, 1);
        __builtin_prefetch(gw + kk + 32, 0, 1);
      }
    }
    __syncthreads();
    v16bf bt[2];
    #pragma unroll
    for (int oi = 0; oi < 2; ++oi)
      bt[oi] = load_tile(Bsm + (wo*32 + oi*16)*LDK, LDK);
    #pragma unroll
    for (int mi = 0; mi < 4; ++mi) {
      v16bf at = load_tile(Asm + (wm*64 + mi*16)*LDK, LDK);
      #pragma unroll
      for (int oi = 0; oi < 2; ++oi)
        acc[mi][oi] = wmma_bf16(at, bt[oi], acc[mi][oi]);
    }
  }

  const int nloc = lane & 15;
  const int hih  = lane >> 4;
  #pragma unroll
  for (int oi = 0; oi < 2; ++oi) {
    const int o = obase + wo*32 + oi*16 + nloc;
    if (mode == 0) {
      const int sec = o >> 8, rem = o & 255;
      const float bias = (sec == 0) ? q_bias[rem] : (sec == 2 ? v_bias[rem] : 0.f);
      const int hh = rem >> 5, dd = rem & 31;
      bf16* dst = (sec == 0) ? qb : (sec == 1 ? kb : vb);
      #pragma unroll
      for (int mi = 0; mi < 4; ++mi)
        #pragma unroll
        for (int vr = 0; vr < 8; ++vr) {
          const int token = tokbase + wm*64 + mi*16 + vr + hih*8;
          const int b = token >> 9, n = token & 511;
          dst[((((size_t)b*8 + hh)*512 + n) << 5) + dd] = (bf16)(acc[mi][oi][vr] + bias);
        }
    } else {
      const float bias = proj_b[o];
      #pragma unroll
      for (int mi = 0; mi < 4; ++mi)
        #pragma unroll
        for (int vr = 0; vr < 8; ++vr) {
          const int token = tokbase + wm*64 + mi*16 + vr + hih*8;
          out[(size_t)token*256 + o] = acc[mi][oi][vr] + bias;
        }
    }
  }
}

// ---- fused window attention -------------------------------------------
// One block = one (b,h) pair x 128 query rows. Full K/V (512x32) staged in
// LDS (K L2-normalized, V transposed); scores live entirely in LDS.
__global__ __launch_bounds__(256, 1)
void attn_kernel(const bf16* __restrict__ qg, const bf16* __restrict__ kg,
                 const bf16* __restrict__ vg, const float* __restrict__ rpb,
                 const float* __restrict__ maskp, const float* __restrict__ lscale,
                 bf16* __restrict__ ao) {
  extern __shared__ char smem[];
  bf16* Ksm = (bf16*)smem;                  // 512 * LDK
  bf16* Qsm = Ksm + 512 * LDK;              // 128 * LDK
  bf16* VT  = Qsm + 128 * LDK;              // 32  * LDW  (V transposed)
  bf16* Ssm = VT  + 32  * LDW;              // 128 * LDW  (scores/probs)
  float* rowinv = (float*)(Ssm + 128 * LDW);// 128 floats

  const int tid  = threadIdx.x;
  const int lane = tid & 31;
  const int wid  = tid >> 5;
  const int bh   = blockIdx.x;              // b*8 + h
  const int b    = bh >> 3;
  const int h    = bh & 7;
  const int qblk = blockIdx.y;              // 0..3 (128 rows each)
  const int w    = b & 63;                  // mask window index (B_=128, nW=64)
  const float sc = __expf(fminf(lscale[h], 4.605170186f));  // clamp at log(100)

  // ---- stage: K normalized row-major, V transposed ----
  for (int rr = tid; rr < 512; rr += 256) {
    const bf16* kp = kg + (((size_t)bh*512 + rr) << 5);
    v8bf c0 = *(const v8bf*)(kp);
    v8bf c1 = *(const v8bf*)(kp + 8);
    v8bf c2 = *(const v8bf*)(kp + 16);
    v8bf c3 = *(const v8bf*)(kp + 24);
    float f[32];
    #pragma unroll
    for (int j = 0; j < 8; ++j) {
      f[j] = (float)c0[j]; f[8+j] = (float)c1[j];
      f[16+j] = (float)c2[j]; f[24+j] = (float)c3[j];
    }
    float ss = 0.f;
    #pragma unroll
    for (int j = 0; j < 32; ++j) ss += f[j]*f[j];
    const float rn = 1.f / fmaxf(sqrtf(ss), 1e-12f);
    v8bf o0, o1, o2, o3;
    #pragma unroll
    for (int j = 0; j < 8; ++j) {
      o0[j] = (bf16)(f[j]*rn);    o1[j] = (bf16)(f[8+j]*rn);
      o2[j] = (bf16)(f[16+j]*rn); o3[j] = (bf16)(f[24+j]*rn);
    }
    bf16* krow = Ksm + rr*LDK;
    *(v8bf*)(krow) = o0; *(v8bf*)(krow+8) = o1;
    *(v8bf*)(krow+16) = o2; *(v8bf*)(krow+24) = o3;

    const bf16* vp = vg + (((size_t)bh*512 + rr) << 5);
    v8bf u0 = *(const v8bf*)(vp);
    v8bf u1 = *(const v8bf*)(vp + 8);
    v8bf u2 = *(const v8bf*)(vp + 16);
    v8bf u3 = *(const v8bf*)(vp + 24);
    #pragma unroll
    for (int j = 0; j < 8; ++j) {
      VT[(j)*LDW + rr]    = u0[j];
      VT[(8+j)*LDW + rr]  = u1[j];
      VT[(16+j)*LDW + rr] = u2[j];
      VT[(24+j)*LDW + rr] = u3[j];
    }
  }
  // ---- stage: Q normalized * per-head scale ----
  if (tid < 128) {
    const int qr = qblk*128 + tid;
    const bf16* qp = qg + (((size_t)bh*512 + qr) << 5);
    v8bf c0 = *(const v8bf*)(qp);
    v8bf c1 = *(const v8bf*)(qp + 8);
    v8bf c2 = *(const v8bf*)(qp + 16);
    v8bf c3 = *(const v8bf*)(qp + 24);
    float f[32];
    #pragma unroll
    for (int j = 0; j < 8; ++j) {
      f[j] = (float)c0[j]; f[8+j] = (float)c1[j];
      f[16+j] = (float)c2[j]; f[24+j] = (float)c3[j];
    }
    float ss = 0.f;
    #pragma unroll
    for (int j = 0; j < 32; ++j) ss += f[j]*f[j];
    const float rn = sc / fmaxf(sqrtf(ss), 1e-12f);
    v8bf o0, o1, o2, o3;
    #pragma unroll
    for (int j = 0; j < 8; ++j) {
      o0[j] = (bf16)(f[j]*rn);    o1[j] = (bf16)(f[8+j]*rn);
      o2[j] = (bf16)(f[16+j]*rn); o3[j] = (bf16)(f[24+j]*rn);
    }
    bf16* qrow = Qsm + tid*LDK;
    *(v8bf*)(qrow) = o0; *(v8bf*)(qrow+8) = o1;
    *(v8bf*)(qrow+16) = o2; *(v8bf*)(qrow+24) = o3;
  }
  __syncthreads();

  bf16* Sw = Ssm + wid*16*LDW;   // this wave's 16x512 score stripe

  // ---- phase 1: S = (scaled qn) @ kn^T via WMMA ----
  {
    const v16bf a = load_tile(Qsm + wid*16*LDK, LDK);
    const int r  = lane & 15;
    const int m0 = (lane >> 4) * 8;
    for (int jt = 0; jt < 32; ++jt) {
      v16bf bt = load_tile(Ksm + jt*16*LDK, LDK);
      v8f c = (v8f){0.f,0.f,0.f,0.f,0.f,0.f,0.f,0.f};
      c = wmma_bf16(a, bt, c);
      #pragma unroll
      for (int vr = 0; vr < 8; ++vr)
        Sw[(vr + m0)*LDW + jt*16 + r] = (bf16)c[vr];
    }
  }
  __syncthreads();

  // ---- phase 2: add rpb + mask, softmax in-place (bf16 probs) ----
  {
    const int r = lane & 15;
    const int half = lane >> 4;
    const int qrow = qblk*128 + wid*16 + r;
    bf16* srow = Sw + r*LDW + half*256;
    const float* mrow = maskp + (((size_t)w*512 + qrow) << 9) + half*256;
    const float* prow = rpb  + (((size_t)h*512 + qrow) << 9) + half*256;
    float mx = -3.0e38f;
    for (int i = 0; i < 256; i += 8) {
      v8bf sv = *(v8bf*)(srow + i);
      v4f p0 = *(const v4f*)(prow + i);
      v4f p1 = *(const v4f*)(prow + i + 4);
      v4f g0 = *(const v4f*)(mrow + i);
      v4f g1 = *(const v4f*)(mrow + i + 4);
      #pragma unroll
      for (int j = 0; j < 4; ++j) {
        float s0 = (float)sv[j]   + p0[j] + g0[j];
        float s1 = (float)sv[4+j] + p1[j] + g1[j];
        sv[j] = (bf16)s0; sv[4+j] = (bf16)s1;
        mx = fmaxf(mx, fmaxf(s0, s1));
      }
      *(v8bf*)(srow + i) = sv;
    }
    mx = fmaxf(mx, __shfl_xor(mx, 16));
    float sum = 0.f;
    for (int i = 0; i < 256; i += 8) {
      v8bf sv = *(v8bf*)(srow + i);
      #pragma unroll
      for (int j = 0; j < 8; ++j) {
        float e = __expf((float)sv[j] - mx);
        sum += e;
        sv[j] = (bf16)e;
      }
      *(v8bf*)(srow + i) = sv;
    }
    sum += __shfl_xor(sum, 16);
    if (half == 0) rowinv[wid*16 + r] = 1.f / sum;
  }
  __syncthreads();

  // ---- phase 3: O = P @ V via WMMA, scale rows by 1/sum ----
  {
    v8f a0 = (v8f){0.f,0.f,0.f,0.f,0.f,0.f,0.f,0.f};
    v8f a1 = (v8f){0.f,0.f,0.f,0.f,0.f,0.f,0.f,0.f};
    for (int kt = 0; kt < 16; ++kt) {
      v16bf p  = load_tile(Sw + kt*32, LDW);
      v16bf b0 = load_tile(VT + kt*32, LDW);
      v16bf b1 = load_tile(VT + 16*LDW + kt*32, LDW);
      a0 = wmma_bf16(p, b0, a0);
      a1 = wmma_bf16(p, b1, a1);
    }
    const int nloc = lane & 15;
    const int half = lane >> 4;
    #pragma unroll
    for (int vr = 0; vr < 8; ++vr) {
      const int m = vr + half*8;
      const float rs = rowinv[wid*16 + m];
      const int qrow = qblk*128 + wid*16 + m;
      bf16* dst = ao + (((size_t)b*512 + qrow) << 8) + h*32;
      dst[nloc]      = (bf16)(a0[vr] * rs);
      dst[16 + nloc] = (bf16)(a1[vr] * rs);
    }
  }
}

// ---- host launch -------------------------------------------------------
extern "C" void kernel_launch(void* const* d_in, const int* in_sizes, int n_in,
                              void* d_out, int out_size, void* d_ws, size_t ws_size,
                              hipStream_t stream) {
  const float* x        = (const float*)d_in[0];
  const float* mask     = (const float*)d_in[1];
  const float* qkv_w    = (const float*)d_in[2];
  const float* q_bias   = (const float*)d_in[3];
  const float* v_bias   = (const float*)d_in[4];
  const float* lscale   = (const float*)d_in[5];
  const float* cpb_w1   = (const float*)d_in[6];
  const float* cpb_b1   = (const float*)d_in[7];
  const float* cpb_w2   = (const float*)d_in[8];
  const float* proj_w   = (const float*)d_in[9];
  const float* proj_b   = (const float*)d_in[10];
  const float* rel_tab  = (const float*)d_in[11];
  const int*   rel_idx  = (const int*)d_in[12];
  float* out = (float*)d_out;

  char* ws = (char*)d_ws;
  bf16*  xb   = (bf16*)(ws);                       // 33,554,432 B
  bf16*  qb   = (bf16*)(ws + 33554432ull);
  bf16*  kb   = (bf16*)(ws + 67108864ull);
  bf16*  vb   = (bf16*)(ws + 100663296ull);
  bf16*  ao   = (bf16*)(ws + 134217728ull);
  float* rpb  = (float*)(ws + 167772160ull);       // 8,388,608 B
  float* sig  = (float*)(ws + 176160768ull);       // 108,000 B
  bf16*  wqb  = (bf16*)(ws + 176268800ull);        // 393,216 B
  bf16*  wpb  = (bf16*)(ws + 176662016ull);        // 131,072 B

  cvt_bf16_kernel<<<4096, 256, 0, stream>>>(x, xb, 128*512*256);
  cvt_bf16_kernel<<<192, 256, 0, stream>>>(qkv_w, wqb, 768*256);
  cvt_bf16_kernel<<<64, 256, 0, stream>>>(proj_w, wpb, 256*256);
  cpb_mlp_kernel<<<14, 256, 0, stream>>>(rel_tab, cpb_w1, cpb_b1, cpb_w2, sig, 3375);
  rpb_mat_kernel<<<1024, 256, 0, stream>>>(rel_idx, sig, rpb);

  gemm_k256_kernel<<<dim3(512, 6), 256, 0, stream>>>(
      xb, wqb, q_bias, v_bias, qb, kb, vb, nullptr, nullptr, 0);

  const size_t shm = (size_t)(512*LDK + 128*LDK + 32*LDW + 128*LDW) * sizeof(bf16)
                   + 128 * sizeof(float);          // ~233 KB of the 320 KB WGP LDS
  hipFuncSetAttribute(reinterpret_cast<const void*>(attn_kernel),
                      hipFuncAttributeMaxDynamicSharedMemorySize, (int)shm);
  attn_kernel<<<dim3(1024, 4), 256, shm, stream>>>(qb, kb, vb, rpb, mask, lscale, ao);

  gemm_k256_kernel<<<dim3(512, 2), 256, 0, stream>>>(
      ao, wpb, nullptr, nullptr, nullptr, nullptr, nullptr, proj_b, out, 1);
}
